// EncoderLayer_20581483282578
// MI455X (gfx1250) — matrix operand order repaired and checked by reading, not compiled, not run on hardware
//
#include <hip/hip_runtime.h>
#include <hip/hip_bf16.h>
#include <math.h>

// ---------------------------------------------------------------------------
// Types for CDNA5 WMMA
// ---------------------------------------------------------------------------
typedef __bf16 bf16_t;
typedef __attribute__((ext_vector_type(16))) __bf16 v16bf;
typedef __attribute__((ext_vector_type(8)))  float  v8f;

static __device__ __forceinline__ bf16_t f2bf(float f) {
  unsigned u = __builtin_bit_cast(unsigned, f);
  u += 0x7FFFu + ((u >> 16) & 1u);          // round-to-nearest-even
  unsigned short s = (unsigned short)(u >> 16);
  return __builtin_bit_cast(bf16_t, s);
}

// ---------------------------------------------------------------------------
// Problem constants
// ---------------------------------------------------------------------------
#define MS_B  4
#define MS_LV 5440
#define MS_NH 8
#define MS_D  256
#define MS_M  (MS_B * MS_LV)        // 21760

// ---------------------------------------------------------------------------
// A-fragment loader from LDS (16x32 bf16, MxK). Per CDNA5 ISA 7.12.2:
// lane 0-15: M=lane, K-halves {0..7,16..23}; lane 16-31: K-halves {8..15,24..31}.
// Compiles to two ds_load_b128 per fragment.
// ---------------------------------------------------------------------------
#define LDSA_STRIDE 40   // bf16 elems; 80B rows keep 16B store alignment

static __device__ __forceinline__ v16bf load_fragA(const bf16_t* sA,
                                                   int mBase, int lane) {
  int r  = mBase + (lane & 15);
  int kh = (lane >> 4) & 1;
  const bf16_t* row = sA + r * LDSA_STRIDE;
  v16bf f;
#pragma unroll
  for (int i = 0; i < 8; ++i) {
    int kk = ((i & 4) << 2) + kh * 8 + ((i & 3) << 1);
    f[2 * i]     = row[kk];
    f[2 * i + 1] = row[kk + 1];
  }
  return f;
}

// ---------------------------------------------------------------------------
// Weight pre-shuffle: W (KxN fp32, row-major) -> fragment-ordered bf16 so a
// wave can load a whole 32x16 B-fragment as one contiguous 32B/lane read.
// Fragment element (lane, e):  n = ntile*16 + (lane&15),
//                              k = kb*32 + (e>>3)*16 + (lane>>4)*8 + (e&7).
// Linear layout: out[(((ntile*KB + kb)*32 + lane)*16) + e]
// ---------------------------------------------------------------------------
__global__ __launch_bounds__(256) void shuffle_weight_kernel(
    const float* __restrict__ W, bf16_t* __restrict__ out, int K, int N) {
  const int idx = blockIdx.x * 256 + threadIdx.x;
  if (idx >= K * N) return;
  const int KB    = K >> 5;
  const int e     = idx & 15;
  const int lane  = (idx >> 4) & 31;
  const int rest  = idx >> 9;
  const int kb    = rest % KB;
  const int ntile = rest / KB;
  const int n = ntile * 16 + (lane & 15);
  const int k = kb * 32 + (e >> 3) * 16 + ((lane >> 4) & 1) * 8 + (e & 7);
  out[idx] = f2bf(W[(size_t)k * N + n]);
}

// fp32 -> bf16 elementwise (for GEMM A inputs produced outside our kernels)
__global__ __launch_bounds__(256) void cvt_bf16_kernel(
    const float* __restrict__ x, bf16_t* __restrict__ y, int n) {
  const int i = blockIdx.x * 256 + threadIdx.x;
  if (i < n) y[i] = f2bf(x[i]);
}

// ---------------------------------------------------------------------------
// WMMA GEMM:  C[M,N] = act( A[M,K] * B[K,N] + bias[N] )
// A: bf16 row-major (staged through LDS, 16B copies, no conversion).
// B: fragment-ordered bf16 in global (L2-resident, 32B/lane direct loads).
// Block: 256 threads (8 waves), tile 64 x (64*NT); wave tile 32 x (16*NT)
// -> 2*NT v_wmma per K-step vs 2 LDS A-fragment loads.
// RELU / HASF (fp32 out) / HASB (bf16 out) are compile-time -> branch-free
// epilogue. Requires M%64==0, N%(64*NT)==0, K%32==0 (true for all calls).
// ---------------------------------------------------------------------------
template <int NT, bool RELU, bool HASF, bool HASB>
__global__ __launch_bounds__(256) void gemm_wmma_kernel(
    const bf16_t* __restrict__ Ab, const bf16_t* __restrict__ Bw,
    const float* __restrict__ bias,
    float* __restrict__ Cf,          // fp32 out (used iff HASF)
    bf16_t* __restrict__ Cb,         // bf16 out (used iff HASB)
    int M, int N, int K) {
  __shared__ bf16_t sA[64 * LDSA_STRIDE];

  const int tid  = threadIdx.x;
  const int lane = tid & 31;
  const int wave = tid >> 5;
  const int bm   = blockIdx.x * 64;
  const int bn   = blockIdx.y * (64 * NT);
  const int waveM  = (wave & 1) * 32;
  const int nTile0 = (bn >> 4) + (wave >> 1) * NT;   // global 16-wide n-tile
  const int KB = K >> 5;

  const int stR = tid >> 2;             // A-stage row  (0..63)
  const int stC = (tid & 3) * 8;        // A-stage col  (0,8,16,24)
  const bf16_t* aSrc = Ab + (size_t)(bm + stR) * K + stC;
  bf16_t* aDst = sA + stR * LDSA_STRIDE + stC;
  const v16bf* bFrag = (const v16bf*)Bw;

  v8f acc[2][NT] = {};

  for (int kb = 0; kb < KB; ++kb) {
    // Stage A tile (64x32 bf16 = 4KB): one 16B vector copy per thread.
    *(uint4*)aDst = *(const uint4*)(aSrc + (size_t)kb * 32);
    if (kb + 1 < KB)
      __builtin_prefetch(aSrc + (size_t)(kb + 1) * 32, 0, 0);
    __syncthreads();

    v16bf aF[2], bF[NT];
    aF[0] = load_fragA(sA, waveM, lane);
    aF[1] = load_fragA(sA, waveM + 16, lane);
#pragma unroll
    for (int nt = 0; nt < NT; ++nt)
      bF[nt] = bFrag[((size_t)(nTile0 + nt) * KB + kb) * 32 + lane];

#pragma unroll
    for (int mi = 0; mi < 2; ++mi)
#pragma unroll
      for (int nt = 0; nt < NT; ++nt)
        acc[mi][nt] = __builtin_amdgcn_wmma_f32_16x16x32_bf16(
            false, aF[mi], false, bF[nt], (short)0, acc[mi][nt], false, false);

    __syncthreads();
  }

  // Epilogue. C/D layout: lane = N%16, VGPR v = M%8, lane>=16 => M+8.
  const int nIn = lane & 15;
  const int mHi = (lane >> 4) * 8;
#pragma unroll
  for (int mi = 0; mi < 2; ++mi) {
#pragma unroll
    for (int nt = 0; nt < NT; ++nt) {
      const int gn = (nTile0 + nt) * 16 + nIn;
      const float bv = bias[gn];
#pragma unroll
      for (int v = 0; v < 8; ++v) {
        const int gm = bm + waveM + 16 * mi + mHi + v;
        float x = acc[mi][nt][v] + bv;
        if (RELU) x = fmaxf(x, 0.0f);
        if (HASF) Cf[(size_t)gm * N + gn] = x;
        if (HASB) Cb[(size_t)gm * N + gn] = f2bf(x);
      }
    }
  }
}

// ---------------------------------------------------------------------------
// MSDA sampling: softmax(16) + bilinear gather + weighted accumulate.
// One wave per (b, q, h); lane = channel within head (DH == 32 == wave32).
// Each corner gather is one coalesced 128B line; v (22MB) is L2-resident.
// Output written directly as bf16 (feeds the W_out GEMM).
// ---------------------------------------------------------------------------
__global__ __launch_bounds__(256) void msda_sample_kernel(
    const float* __restrict__ v, const float* __restrict__ off,
    const float* __restrict__ logits, const float* __restrict__ refs,
    bf16_t* __restrict__ sampb) {
  const int lsi[4] = {0, 4096, 5120, 5376};
  const int Hs[4]  = {64, 32, 16, 8};
  const int Ws[4]  = {64, 32, 16, 8};

  const int lane = threadIdx.x & 31;
  const int wave = threadIdx.x >> 5;
  const int item = blockIdx.x * 8 + wave;     // < B*LV*NH = 174080
  const int h  = item & 7;
  const int t  = item >> 3;
  const int qi = t % MS_LV;
  const int b  = t / MS_LV;
  const long row = (long)b * MS_LV + qi;

  // softmax over 16 logits (redundant per lane; contiguous 64B)
  const float* lg = logits + row * 128 + h * 16;
  float p[16];
  float mx = -INFINITY;
#pragma unroll
  for (int i = 0; i < 16; ++i) mx = fmaxf(mx, lg[i]);
  float s = 0.f;
#pragma unroll
  for (int i = 0; i < 16; ++i) { p[i] = __expf(lg[i] - mx); s += p[i]; }
  const float inv = 1.0f / s;

  const float* offp = off + row * MS_D + h * 32;   // (h*NL+l)*NP*2 block
  const float* refb = refs + row * 8;              // [l][2]
  float acc = 0.f;

#pragma unroll
  for (int l = 0; l < 4; ++l) {
    const int   Wl = Ws[l], Hl = Hs[l];
    const float Wf = (float)Wl, Hf = (float)Hl;
    const float rx = refb[l * 2 + 0];
    const float ry = refb[l * 2 + 1];
    const float* vbase =
        v + ((long)b * MS_LV + lsi[l]) * MS_D + h * 32 + lane;
#pragma unroll
    for (int pp = 0; pp < 4; ++pp) {
      const float ox = offp[(l * 4 + pp) * 2 + 0];
      const float oy = offp[(l * 4 + pp) * 2 + 1];
      const float x = rx * Wf + ox - 0.5f;   // (rx + ox/W)*W - 0.5
      const float y = ry * Hf + oy - 0.5f;
      const float x0 = floorf(x), y0 = floorf(y);
      const float tx = x - x0,   ty = y - y0;
      const int xi0 = (int)x0, yi0 = (int)y0;
      const float aw = p[l * 4 + pp] * inv;

      const float w00 = (1.f - tx) * (1.f - ty);
      const float w10 = tx * (1.f - ty);
      const float w01 = (1.f - tx) * ty;
      const float w11 = tx * ty;

      if (xi0 >= 0 && xi0 < Wl && yi0 >= 0 && yi0 < Hl)
        acc += aw * w00 * vbase[(size_t)(yi0 * Wl + xi0) * MS_D];
      if (xi0 + 1 >= 0 && xi0 + 1 < Wl && yi0 >= 0 && yi0 < Hl)
        acc += aw * w10 * vbase[(size_t)(yi0 * Wl + xi0 + 1) * MS_D];
      if (xi0 >= 0 && xi0 < Wl && yi0 + 1 >= 0 && yi0 + 1 < Hl)
        acc += aw * w01 * vbase[(size_t)((yi0 + 1) * Wl + xi0) * MS_D];
      if (xi0 + 1 >= 0 && xi0 + 1 < Wl && yi0 + 1 >= 0 && yi0 + 1 < Hl)
        acc += aw * w11 * vbase[(size_t)((yi0 + 1) * Wl + xi0 + 1) * MS_D];
    }
  }
  sampb[row * MS_D + h * 32 + lane] = f2bf(acc);
}

// ---------------------------------------------------------------------------
// Row LayerNorm (D=256), one wave per row, 8 rows per block.
// MODE 0: out = res + LN(x)   (post-MSDA residual; also emits bf16 for FFN)
// MODE 1: out = LN(x + res)   (final LN of q + FFN)
// ---------------------------------------------------------------------------
template <int MODE, bool HASB>
__global__ __launch_bounds__(256) void add_ln_kernel(
    const float* __restrict__ x, const float* __restrict__ res,
    const float* __restrict__ g, const float* __restrict__ bta,
    float* __restrict__ out, bf16_t* __restrict__ outb) {
  const int lane = threadIdx.x & 31;
  const int wave = threadIdx.x >> 5;
  const long row = (long)blockIdx.x * 8 + wave;
  const float* xr = x + row * 256;
  const float* rr = res + row * 256;

  float vals[8];
  float sum = 0.f;
#pragma unroll
  for (int i = 0; i < 8; ++i) {
    float t = xr[lane + i * 32];
    if (MODE == 1) t += rr[lane + i * 32];
    vals[i] = t;
    sum += t;
  }
#pragma unroll
  for (int o = 16; o > 0; o >>= 1) sum += __shfl_xor(sum, o, 32);
  const float mu = sum * (1.0f / 256.0f);

  float vs = 0.f;
#pragma unroll
  for (int i = 0; i < 8; ++i) { float d = vals[i] - mu; vs += d * d; }
#pragma unroll
  for (int o = 16; o > 0; o >>= 1) vs += __shfl_xor(vs, o, 32);
  const float rstd = rsqrtf(vs * (1.0f / 256.0f) + 1e-5f);

#pragma unroll
  for (int i = 0; i < 8; ++i) {
    const int c = lane + i * 32;
    float y = (vals[i] - mu) * rstd * g[c] + bta[c];
    if (MODE == 0) y += rr[c];
    out[row * 256 + c] = y;
    if (HASB) outb[row * 256 + c] = f2bf(y);
  }
}

// ---------------------------------------------------------------------------
// Launch
// ---------------------------------------------------------------------------
extern "C" void kernel_launch(void* const* d_in, const int* in_sizes, int n_in,
                              void* d_out, int out_size, void* d_ws,
                              size_t ws_size, hipStream_t stream) {
  (void)in_sizes; (void)n_in; (void)out_size; (void)ws_size;
  const int M = MS_M;  // 21760

  const float* query  = (const float*)d_in[0];
  const float* refs   = (const float*)d_in[1];
  const float* value  = (const float*)d_in[2];
  // d_in[3] spatial_shapes, d_in[4] level_start_index: compile-time constants
  const float* W_off  = (const float*)d_in[5];
  const float* b_off  = (const float*)d_in[6];
  const float* W_attn = (const float*)d_in[7];
  const float* b_attn = (const float*)d_in[8];
  const float* W_val  = (const float*)d_in[9];
  const float* b_val  = (const float*)d_in[10];
  const float* W_out  = (const float*)d_in[11];
  const float* b_out  = (const float*)d_in[12];
  const float* ln1_g  = (const float*)d_in[13];
  const float* ln1_b  = (const float*)d_in[14];
  const float* ln2_g  = (const float*)d_in[15];
  const float* ln2_b  = (const float*)d_in[16];
  const float* W_ff1  = (const float*)d_in[17];
  const float* b_ff1  = (const float*)d_in[18];
  const float* W_ff2  = (const float*)d_in[19];
  const float* b_ff2  = (const float*)d_in[20];

  // -------- workspace carve (all chunks 32B-aligned) --------
  char* p = (char*)d_ws;
  float* v    = (float*)p;  p += (size_t)M * 256 * 4;   // value proj (fp32)
  float* off  = (float*)p;  p += (size_t)M * 256 * 4;   // offsets (fp32)
  float* attn = (float*)p;  p += (size_t)M * 128 * 4;   // attn logits
  float* msda = (float*)p;  p += (size_t)M * 256 * 4;   // W_out proj
  float* q    = (float*)p;  p += (size_t)M * 256 * 4;   // residual stream
  float* f2   = (float*)p;  p += (size_t)M * 256 * 4;   // FFN down proj
  bf16_t* valb  = (bf16_t*)p; p += (size_t)M * 256 * 2; // bf16 activations
  bf16_t* qryb  = (bf16_t*)p; p += (size_t)M * 256 * 2;
  bf16_t* sampb = (bf16_t*)p; p += (size_t)M * 256 * 2;
  bf16_t* qb    = (bf16_t*)p; p += (size_t)M * 256 * 2;
  bf16_t* h1b   = (bf16_t*)p; p += (size_t)M * 1024 * 2;
  bf16_t* sWval = (bf16_t*)p; p += (size_t)256 * 256 * 2;  // shuffled weights
  bf16_t* sWoff = (bf16_t*)p; p += (size_t)256 * 256 * 2;
  bf16_t* sWatt = (bf16_t*)p; p += (size_t)256 * 128 * 2;
  bf16_t* sWout = (bf16_t*)p; p += (size_t)256 * 256 * 2;
  bf16_t* sWff1 = (bf16_t*)p; p += (size_t)256 * 1024 * 2;
  bf16_t* sWff2 = (bf16_t*)p; p += (size_t)1024 * 256 * 2;

  const dim3 blk(256);
  const int MB = M / 64;  // 340

  // -------- prep: bf16 activations + fragment-ordered weights --------
  cvt_bf16_kernel<<<dim3(M * 256 / 256), blk, 0, stream>>>(value, valb, M * 256);
  cvt_bf16_kernel<<<dim3(M * 256 / 256), blk, 0, stream>>>(query, qryb, M * 256);
  shuffle_weight_kernel<<<dim3(256 * 256 / 256), blk, 0, stream>>>(W_val, sWval, 256, 256);
  shuffle_weight_kernel<<<dim3(256 * 256 / 256), blk, 0, stream>>>(W_off, sWoff, 256, 256);
  shuffle_weight_kernel<<<dim3(256 * 128 / 256), blk, 0, stream>>>(W_attn, sWatt, 256, 128);
  shuffle_weight_kernel<<<dim3(256 * 256 / 256), blk, 0, stream>>>(W_out, sWout, 256, 256);
  shuffle_weight_kernel<<<dim3(256 * 1024 / 256), blk, 0, stream>>>(W_ff1, sWff1, 256, 1024);
  shuffle_weight_kernel<<<dim3(1024 * 256 / 256), blk, 0, stream>>>(W_ff2, sWff2, 1024, 256);

  // -------- input projections --------
  gemm_wmma_kernel<4, false, true, false><<<dim3(MB, 1), blk, 0, stream>>>(
      valb, sWval, b_val, v, nullptr, M, 256, 256);
  gemm_wmma_kernel<4, false, true, false><<<dim3(MB, 1), blk, 0, stream>>>(
      qryb, sWoff, b_off, off, nullptr, M, 256, 256);
  gemm_wmma_kernel<2, false, true, false><<<dim3(MB, 1), blk, 0, stream>>>(
      qryb, sWatt, b_attn, attn, nullptr, M, 128, 256);

  // -------- softmax + bilinear sampling --------
  msda_sample_kernel<<<dim3(M * MS_NH / 8), blk, 0, stream>>>(
      v, off, attn, refs, sampb);

  // -------- output projection, residual + LN1 --------
  gemm_wmma_kernel<4, false, true, false><<<dim3(MB, 1), blk, 0, stream>>>(
      sampb, sWout, b_out, msda, nullptr, M, 256, 256);
  add_ln_kernel<0, true><<<dim3(M / 8), blk, 0, stream>>>(
      msda, value, ln1_g, ln1_b, q, qb);

  // -------- FFN --------
  gemm_wmma_kernel<4, true, false, true><<<dim3(MB, 4), blk, 0, stream>>>(
      qb, sWff1, b_ff1, nullptr, h1b, M, 1024, 256);
  gemm_wmma_kernel<4, false, true, false><<<dim3(MB, 1), blk, 0, stream>>>(
      h1b, sWff2, b_ff2, f2, nullptr, M, 256, 1024);

  // -------- final residual + LN2 -> d_out --------
  add_ln_kernel<1, false><<<dim3(M / 8), blk, 0, stream>>>(
      q, f2, ln2_g, ln2_b, (float*)d_out, nullptr);
}
